// CellularNN_36687610642996
// MI455X (gfx1250) — matrix operand order repaired and checked by reading, not compiled, not run on hardware
//
#include <hip/hip_runtime.h>
#include <hip/hip_bf16.h>

// CDNA5 / gfx1250, wave32. Fused NCA step:
//   perceive -> (112->128 relu -> 32) 1x1 MLP via v_wmma_f32_16x16x32_f16
//   -> residual -> argmax hidden mask -> channel LayerNorm.
// One workgroup (256 thr = 8 waves) per (batch, 4-row band); one wave per 16
// pixels of a row. Weights staged to LDS (f16) once per workgroup and the
// GEMM2 A-fragments are hoisted into registers across the row loop.

typedef __attribute__((ext_vector_type(16))) _Float16 v16h;
typedef __attribute__((ext_vector_type(8)))  float    v8f;

#define BATCH 64
#define CCH   32      // state channels
#define HDIM  128
#define WDIM  128
#define NCLS  10
#define PERC  112     // 32 + 10*8
#define KPAD  128     // PERC padded to multiple of 32
#define NHID  128
#define ROWS  4       // rows per workgroup

// ---- A-matrix fragment (16x32 f16), rows m = lane&15, contiguous K:
// lanes 0-15: V0..7 -> K=0..15 ; lanes 16-31: K=16..31.
__device__ __forceinline__ v16h load_a16(const _Float16* w, int ld, int row,
                                         int kblock, int lane) {
  v16h a;
  const int ko = kblock * 32 + ((lane & 16) ? 16 : 0);
  const _Float16* p = w + row * ld + ko;
#pragma unroll
  for (int v = 0; v < 8; ++v) {
    const int kb = 2 * v;
    a[2 * v]     = p[kb];
    a[2 * v + 1] = p[kb + 1];
  }
  return a;
}

__global__ __launch_bounds__(256) void nca_fused_kernel(
    const float* __restrict__ x,   const float* __restrict__ w1,
    const float* __restrict__ b1,  const float* __restrict__ w2,
    const float* __restrict__ b2,  const float* __restrict__ gamma,
    const float* __restrict__ beta, float* __restrict__ out) {
  __shared__ _Float16 lds_w1[NHID * KPAD];     // 32 KB, K=112..127 zero-padded
  __shared__ _Float16 lds_w2[CCH * NHID];      // 8 KB
  __shared__ float    lds_x [CCH * WDIM];      // 16 KB, full row y (f32 for residual)
  __shared__ _Float16 lds_cls[NCLS * 2 * WDIM];// 5 KB, class rows y-1 / y+1
  __shared__ float    lds_b1[NHID];
  __shared__ float    lds_b2[CCH], lds_g[CCH], lds_be[CCH];

  const int tid   = threadIdx.x;
  const int bi    = blockIdx.x;
  const int b     = bi >> 5;            // / (HDIM/ROWS)
  const int ybase = (bi & 31) * ROWS;

  // ---------------- stage weights/biases once ----------------
  for (int i = tid; i < NHID * KPAD; i += 256) {
    const int o = i >> 7, k = i & 127;
    lds_w1[i] = (k < PERC) ? (_Float16)w1[o * PERC + k] : (_Float16)0.0f;
  }
  for (int i = tid; i < CCH * NHID; i += 256) lds_w2[i] = (_Float16)w2[i];
  if (tid < NHID) lds_b1[tid] = b1[tid];
  if (tid < CCH) { lds_b2[tid] = b2[tid]; lds_g[tid] = gamma[tid]; lds_be[tid] = beta[tid]; }
  __syncthreads();

  const int wave = tid >> 5;
  const int lane = tid & 31;
  const int nn   = lane & 15;          // pixel within wave / B,C column
  const int g16  = lane & 16;          // B-layout K half
  const int hh   = g16 ? 8 : 0;        // C-layout row half
  const int px   = wave * 16 + nn;     // pixel x coordinate

  // Hoist GEMM2 A-fragments (w2) across the row loop: 2 m-tiles x 4 k-blocks.
  v16h a2[2][4];
#pragma unroll
  for (int mo = 0; mo < 2; ++mo)
#pragma unroll
    for (int kb = 0; kb < 4; ++kb)
      a2[mo][kb] = load_a16(lds_w2, NHID, mo * 16 + nn, kb, lane);

  for (int ry = 0; ry < ROWS; ++ry) {
    const int y = ybase + ry;

    // ---------------- stage this row ----------------
    for (int i = tid; i < CCH * WDIM; i += 256) {
      const int c = i >> 7, xc = i & 127;
      lds_x[i] = x[(((size_t)b * CCH + c) * HDIM + y) * WDIM + xc];
    }
    for (int i = tid; i < NCLS * 2 * WDIM; i += 256) {
      const int c = i >> 8, r = (i >> 7) & 1, xc = i & 127;
      const int yy = y + (r ? 1 : -1);
      float v = 0.0f;
      if (yy >= 0 && yy < HDIM)
        v = x[(((size_t)b * CCH + c) * HDIM + yy) * WDIM + xc];
      lds_cls[i] = (_Float16)v;
    }
    __syncthreads();

    // Build perception B-fragments: K = kb*32 + g16 + i
    v16h pf[4];
#pragma unroll
    for (int kb = 0; kb < 4; ++kb) {
#pragma unroll
      for (int i = 0; i < 16; ++i) {
        const int k = kb * 32 + g16 + i;
        float val = 0.0f;
        if (k < CCH) {
          val = lds_x[k * WDIM + px];
        } else if (k < PERC) {
          const int idx = k - CCH;
          const int c = idx >> 3;
          const int s = idx & 7;
          const int t = (s < 4) ? s : (s + 1);        // skip center tap
          const int dy = t / 3 - 1;
          const int dx = t % 3 - 1;
          const int xn = px + dx;
          if (xn >= 0 && xn < WDIM) {
            if (dy == 0) val = lds_x[c * WDIM + xn];
            else         val = (float)lds_cls[(c * 2 + (dy > 0 ? 1 : 0)) * WDIM + xn];
          }
        }
        pf[kb][i] = (_Float16)val;
      }
    }

    // GEMM1: h[128x16] = relu(w1 @ perc + b1); relayout C->B via shfl_xor(16).
    v16h hf[4];
#pragma unroll
    for (int kp = 0; kp < 4; ++kp) {    // K-block of GEMM2 = tile pair of GEMM1
      float r0[8], r1[8];
#pragma unroll
      for (int half = 0; half < 2; ++half) {
        const int mt = kp * 2 + half;
        v8f acc;
#pragma unroll
        for (int v = 0; v < 8; ++v) acc[v] = lds_b1[mt * 16 + hh + v];
#pragma unroll
        for (int kb = 0; kb < 4; ++kb) {
          const v16h a = load_a16(lds_w1, KPAD, mt * 16 + nn, kb, lane);
          acc = __builtin_amdgcn_wmma_f32_16x16x32_f16(
              false, a, false, pf[kb], (short)0, acc, false, false);
        }
#pragma unroll
        for (int v = 0; v < 8; ++v) {
          const float rl = fmaxf(acc[v], 0.0f);
          if (half == 0) r0[v] = rl; else r1[v] = rl;
        }
      }
      // low lanes need tile(2kp) rows 0..15; high lanes tile(2kp+1) rows 0..15
#pragma unroll
      for (int v = 0; v < 8; ++v) {
        const float send = g16 ? r0[v] : r1[v];
        const float recv = __shfl_xor(send, 16, 32);
        const float own  = g16 ? r1[v] : r0[v];
        hf[kp][v]     = (_Float16)(g16 ? recv : own);
        hf[kp][8 + v] = (_Float16)(g16 ? own  : recv);
      }
    }

    // GEMM2: dx[32x16] = w2 @ h + b2
    v8f d0, d1;
#pragma unroll
    for (int v = 0; v < 8; ++v) { d0[v] = lds_b2[hh + v]; d1[v] = lds_b2[16 + hh + v]; }
#pragma unroll
    for (int kb = 0; kb < 4; ++kb)
      d0 = __builtin_amdgcn_wmma_f32_16x16x32_f16(
          false, a2[0][kb], false, hf[kb], (short)0, d0, false, false);
#pragma unroll
    for (int kb = 0; kb < 4; ++kb)
      d1 = __builtin_amdgcn_wmma_f32_16x16x32_f16(
          false, a2[1][kb], false, hf[kb], (short)0, d1, false, false);

    // ---------------- epilogue ----------------
    // lane n holds chans hh..hh+7 and 16+hh..16+hh+7 of pixel px; partner = lane^16.
    float s[16];
#pragma unroll
    for (int v = 0; v < 8; ++v) {
      s[v]     = lds_x[(hh + v) * WDIM + px]      + d0[v];
      s[8 + v] = lds_x[(16 + hh + v) * WDIM + px] + d1[v];
    }

    // argmax over class channels 0..9 (first-max wins)
    float mx; int mi;
    if (hh == 0) {
      mx = s[0]; mi = 0;
#pragma unroll
      for (int v = 1; v < 8; ++v) if (s[v] > mx) { mx = s[v]; mi = v; }
    } else {
      mx = s[0]; mi = 8;
      if (s[1] > mx) { mx = s[1]; mi = 9; }
    }
    const float pmx = __shfl_xor(mx, 16, 32);
    const int   pmi = __shfl_xor(mi, 16, 32);
    if (pmx > mx || (pmx == mx && pmi < mi)) { mx = pmx; mi = pmi; }
    const float inv = (mi != 0) ? 1.0f : 0.0f;

    // zero hidden channels (ch >= 10) where argmax class == 0
#pragma unroll
    for (int v = 0; v < 8; ++v) {
      if (hh + v >= NCLS) s[v] *= inv;  // only hh==8, v>=2
      s[8 + v] *= inv;                  // chans 16..31 always hidden
    }

    // LayerNorm over 32 channels (biased variance, eps = 1e-5)
    float sm = 0.0f, sq = 0.0f;
#pragma unroll
    for (int j = 0; j < 16; ++j) { sm += s[j]; sq += s[j] * s[j]; }
    sm += __shfl_xor(sm, 16, 32);
    sq += __shfl_xor(sq, 16, 32);
    const float mean = sm * (1.0f / 32.0f);
    const float var  = sq * (1.0f / 32.0f) - mean * mean;
    const float rs   = rsqrtf(var + 1e-5f);

#pragma unroll
    for (int v = 0; v < 8; ++v) {
      const int c0 = hh + v, c1 = 16 + hh + v;
      out[(((size_t)b * CCH + c0) * HDIM + y) * WDIM + px] =
          (s[v] - mean) * rs * lds_g[c0] + lds_be[c0];
      out[(((size_t)b * CCH + c1) * HDIM + y) * WDIM + px] =
          (s[8 + v] - mean) * rs * lds_g[c1] + lds_be[c1];
    }

    __syncthreads();   // protect lds_x / lds_cls before next row's restage
  }
}

extern "C" void kernel_launch(void* const* d_in, const int* in_sizes, int n_in,
                              void* d_out, int out_size, void* d_ws, size_t ws_size,
                              hipStream_t stream) {
  (void)in_sizes; (void)n_in; (void)out_size; (void)d_ws; (void)ws_size;
  const float* x     = (const float*)d_in[0];
  const float* w1    = (const float*)d_in[1];
  const float* b1    = (const float*)d_in[2];
  const float* w2    = (const float*)d_in[3];
  const float* b2    = (const float*)d_in[4];
  const float* gamma = (const float*)d_in[5];
  const float* beta  = (const float*)d_in[6];
  float* out = (float*)d_out;

  dim3 grid(BATCH * (HDIM / ROWS));  // 2048 workgroups: one per (batch, 4-row band)
  dim3 block(256);                   // 8 wave32s, 16 pixels each
  nca_fused_kernel<<<grid, block, 0, stream>>>(x, w1, b1, w2, b2, gamma, beta, out);
}
